// differN_new_pca_21285857919123
// MI455X (gfx1250) — matrix operand order repaired and checked by reading, not compiled, not run on hardware
//
#include <hip/hip_runtime.h>
#include <hip/hip_bf16.h>

// ---------------------------------------------------------------------------
// Problem constants
// ---------------------------------------------------------------------------
#define BATCH   4096
#define PIX     1024
#define WIRES   10
#define DSTATE  1024
#define NSTEPS  4
#define LLAYERS 4

typedef __bf16 v16bf __attribute__((ext_vector_type(16)));
typedef __bf16 v8bf  __attribute__((ext_vector_type(8)));
typedef float  v8f   __attribute__((ext_vector_type(8)));
typedef unsigned short us4  __attribute__((ext_vector_type(4)));
typedef unsigned short v16us __attribute__((ext_vector_type(16)));

__device__ __forceinline__ unsigned short f2bf(float f) {
    unsigned int u = __float_as_uint(f);
    unsigned int r = (u + 0x7FFFu + ((u >> 16) & 1u)) >> 16;   // RNE
    return (unsigned short)r;
}

// negate a bf16x16 fragment by flipping sign bits (8x v_xor_b32)
__device__ __forceinline__ v16bf bfneg16(v16bf x) {
    const v16us m = {0x8000u,0x8000u,0x8000u,0x8000u,0x8000u,0x8000u,0x8000u,0x8000u,
                     0x8000u,0x8000u,0x8000u,0x8000u,0x8000u,0x8000u,0x8000u,0x8000u};
    v16us t = __builtin_bit_cast(v16us, x) ^ m;
    return __builtin_bit_cast(v16bf, t);
}

// CZ-ring diagonal sign for basis index p (wire 0 = MSB), range r
__device__ __forceinline__ float czsign(int p, int r) {
    int par = 0;
    for (int j = 0; j < WIRES; ++j) {
        int c = j, t = (j + r) % WIRES;
        par ^= ((p >> (9 - c)) & (p >> (9 - t)) & 1);
    }
    return par ? -1.0f : 1.0f;
}

// ---------------------------------------------------------------------------
// PCA kernels
// ---------------------------------------------------------------------------
__global__ __launch_bounds__(256) void mean_kernel(const float* __restrict__ X,
                                                   float* __restrict__ mu) {
    int p = blockIdx.x * 256 + threadIdx.x;           // 0..1023
    float s = 0.f;
    for (int b = 0; b < BATCH; ++b) s += X[b * PIX + p];
    mu[p] = s * (1.0f / BATCH);
}

__global__ __launch_bounds__(256) void center_kernel(const float* __restrict__ X,
                                                     const float* __restrict__ mu,
                                                     float* __restrict__ Xc) {
    int idx = blockIdx.x * 256 + threadIdx.x;
    int p = idx & (PIX - 1);
    Xc[idx] = X[idx] - mu[p];
}

__global__ __launch_bounds__(256) void cov_kernel(const float* __restrict__ Xc,
                                                  float* __restrict__ C) {
    __shared__ float Xi[32][33];
    __shared__ float Xj[32][33];
    int tid = threadIdx.x, ty = tid >> 4, tx = tid & 15;
    int i0 = blockIdx.y * 32, j0 = blockIdx.x * 32;
    float acc00 = 0, acc01 = 0, acc10 = 0, acc11 = 0;
    for (int b0 = 0; b0 < BATCH; b0 += 32) {
        for (int it = 0; it < 4; ++it) {
            int e = it * 256 + tid, r = e >> 5, c = e & 31;
            Xi[r][c] = Xc[(b0 + r) * PIX + i0 + c];
            Xj[r][c] = Xc[(b0 + r) * PIX + j0 + c];
        }
        __syncthreads();
        for (int kk = 0; kk < 32; ++kk) {
            float a0 = Xi[kk][ty * 2], a1 = Xi[kk][ty * 2 + 1];
            float c0 = Xj[kk][tx * 2], c1 = Xj[kk][tx * 2 + 1];
            acc00 += a0 * c0; acc01 += a0 * c1; acc10 += a1 * c0; acc11 += a1 * c1;
        }
        __syncthreads();
    }
    C[(i0 + ty * 2    ) * PIX + j0 + tx * 2    ] = acc00;
    C[(i0 + ty * 2    ) * PIX + j0 + tx * 2 + 1] = acc01;
    C[(i0 + ty * 2 + 1) * PIX + j0 + tx * 2    ] = acc10;
    C[(i0 + ty * 2 + 1) * PIX + j0 + tx * 2 + 1] = acc11;
}

__global__ __launch_bounds__(256) void initv_kernel(float* __restrict__ V) {
    int idx = blockIdx.x * 256 + threadIdx.x;
    if (idx < PIX * WIRES) V[idx] = __sinf((float)idx * 0.7031f + 0.123f);
}

__global__ __launch_bounds__(256) void matvec_kernel(const float* __restrict__ C,
                                                     const float* __restrict__ V,
                                                     float* __restrict__ W) {
    int idx = blockIdx.x * 256 + threadIdx.x;
    if (idx >= PIX * WIRES) return;
    int p = idx / WIRES, j = idx - p * WIRES;
    float s = 0.f;
    for (int k = 0; k < PIX; ++k) s += C[p * PIX + k] * V[k * WIRES + j];
    W[idx] = s;
}

// Gram-Schmidt: V = orthonormalize(W), 1024 x 10, single block
__global__ __launch_bounds__(256) void gs_kernel(const float* __restrict__ W,
                                                 float* __restrict__ V) {
    __shared__ float red[256];
    __shared__ float rr;
    int tid = threadIdx.x;
    for (int j = 0; j < WIRES; ++j) {
        float wj[4];
        for (int it = 0; it < 4; ++it) wj[it] = W[(tid + it * 256) * WIRES + j];
        for (int k = 0; k < j; ++k) {
            float part = 0.f;
            for (int it = 0; it < 4; ++it) part += V[(tid + it * 256) * WIRES + k] * wj[it];
            red[tid] = part; __syncthreads();
            for (int s = 128; s > 0; s >>= 1) { if (tid < s) red[tid] += red[tid + s]; __syncthreads(); }
            if (tid == 0) rr = red[0];
            __syncthreads();
            float r = rr;
            for (int it = 0; it < 4; ++it) wj[it] -= r * V[(tid + it * 256) * WIRES + k];
            __syncthreads();
        }
        float part = 0.f;
        for (int it = 0; it < 4; ++it) part += wj[it] * wj[it];
        red[tid] = part; __syncthreads();
        for (int s = 128; s > 0; s >>= 1) { if (tid < s) red[tid] += red[tid + s]; __syncthreads(); }
        if (tid == 0) rr = red[0];
        __syncthreads();
        float inv = rsqrtf(rr + 1e-20f);
        for (int it = 0; it < 4; ++it) V[(tid + it * 256) * WIRES + j] = wj[it] * inv;
        __threadfence_block();
        __syncthreads();
    }
}

__global__ __launch_bounds__(256) void feats_kernel(const float* __restrict__ Xc,
                                                    const float* __restrict__ V,
                                                    float* __restrict__ F) {
    int idx = blockIdx.x * 256 + threadIdx.x;
    if (idx >= BATCH * WIRES) return;
    int b = idx / WIRES, j = idx - b * WIRES;
    float s = 0.f;
    for (int p = 0; p < PIX; ++p) s += Xc[b * PIX + p] * V[p * WIRES + j];
    F[idx] = s;
}

__global__ __launch_bounds__(256) void signflip_kernel(const float* __restrict__ F,
                                                       float* __restrict__ signs) {
    __shared__ float bv[256];
    __shared__ int   bi[256];
    int j = blockIdx.x, tid = threadIdx.x;
    float best = -1.f; int besti = 1 << 30;
    for (int b = tid; b < BATCH; b += 256) {
        float v = fabsf(F[b * WIRES + j]);
        if (v > best || (v == best && b < besti)) { best = v; besti = b; }
    }
    bv[tid] = best; bi[tid] = besti; __syncthreads();
    for (int s = 128; s > 0; s >>= 1) {
        if (tid < s) {
            if (bv[tid + s] > bv[tid] || (bv[tid + s] == bv[tid] && bi[tid + s] < bi[tid])) {
                bv[tid] = bv[tid + s]; bi[tid] = bi[tid + s];
            }
        }
        __syncthreads();
    }
    if (tid == 0) {
        float x = F[bi[0] * WIRES + j];
        signs[j] = (x > 0.f) ? 1.f : ((x < 0.f) ? -1.f : 0.f);
    }
}

__global__ __launch_bounds__(256) void applysign_kernel(float* __restrict__ F,
                                                        const float* __restrict__ signs) {
    int idx = blockIdx.x * 256 + threadIdx.x;
    if (idx < BATCH * WIRES) F[idx] *= signs[idx % WIRES];
}

// ---------------------------------------------------------------------------
// Gate / unitary construction
// ---------------------------------------------------------------------------
// gates layout: per (n,i,l) block b = ((n*4+i)*2+l): 4096 floats:
//   A (32x32 complex interleaved, 2048 floats) then B (2048 floats)
__global__ __launch_bounds__(256) void build_gates_kernel(const float* __restrict__ w,
                                                          float* __restrict__ gates) {
    __shared__ float g2[WIRES][2][2][2];
    int blk = blockIdx.x;                  // 0..31
    int n = blk >> 3, i = (blk >> 1) & 3, l = blk & 1;
    int tid = threadIdx.x;
    if (tid < WIRES) {
        int j = tid;
        const float* wp = &w[((((n * LLAYERS + i) * 2 + l) * WIRES + j) * 3)];
        float phi = wp[0], th = wp[1], om = wp[2];
        float ch = __cosf(th * 0.5f), sh = __sinf(th * 0.5f);
        float apo = -(phi + om) * 0.5f;    // angle for e00
        float amo = (phi - om) * 0.5f;     // e01 = -cis(amo)*s ; e10 = cis(-amo)*s
        g2[j][0][0][0] =  __cosf(apo) * ch;  g2[j][0][0][1] =  __sinf(apo) * ch;
        g2[j][0][1][0] = -__cosf(amo) * sh;  g2[j][0][1][1] = -__sinf(amo) * sh;
        g2[j][1][0][0] =  __cosf(-amo) * sh; g2[j][1][0][1] =  __sinf(-amo) * sh;
        g2[j][1][1][0] =  __cosf(-apo) * ch; g2[j][1][1][1] =  __sinf(-apo) * ch;
    }
    __syncthreads();
    for (int it = 0; it < 8; ++it) {
        int e = it * 256 + tid;            // 0..2047
        int isB = e >> 10; int ee = e & 1023;
        int y = ee >> 5, yp = ee & 31;
        float pr = 1.f, pi = 0.f;
        for (int j = 0; j < 5; ++j) {
            int gj = isB ? (5 + j) : j;
            int by = (y >> (4 - j)) & 1, byp = (yp >> (4 - j)) & 1;
            float gr = g2[gj][by][byp][0], gi = g2[gj][by][byp][1];
            float nr = pr * gr - pi * gi, ni = pr * gi + pi * gr;
            pr = nr; pi = ni;
        }
        gates[blk * 4096 + isB * 2048 + ee * 2    ] = pr;
        gates[blk * 4096 + isB * 2048 + ee * 2 + 1] = pi;
    }
}

// Build one column q of layer unitary U as row q of U^T (bf16 Br/Bi),
// or (only_col0) column 0 in f32 -> s0.
__global__ __launch_bounds__(256) void build_u_kernel(const float* __restrict__ g,
                                                      unsigned short* __restrict__ Br,
                                                      unsigned short* __restrict__ Bi,
                                                      float* __restrict__ s0,
                                                      int only_col0) {
    __shared__ float S[32][32][2];
    __shared__ float T[32][32][2];
    __shared__ float A1s[2048];
    __shared__ float B1s[2048];
    int q = blockIdx.x, tid = threadIdx.x;
    int yq = q >> 5, xq = q & 31;
    for (int it = 0; it < 8; ++it) {
        A1s[it * 256 + tid] = g[4096 + it * 256 + tid];
        B1s[it * 256 + tid] = g[6144 + it * 256 + tid];
    }
    // S = outer(A0[:,yq], B0[:,xq]) * cz(r=1)
    for (int it = 0; it < 4; ++it) {
        int e = it * 256 + tid; int y = e >> 5, x = e & 31;
        float ar = g[(y * 32 + yq) * 2], ai = g[(y * 32 + yq) * 2 + 1];
        float br = g[2048 + (x * 32 + xq) * 2], bi = g[2048 + (x * 32 + xq) * 2 + 1];
        float sr = ar * br - ai * bi, si = ar * bi + ai * br;
        float sg = czsign((y << 5) | x, 1);
        S[y][x][0] = sr * sg; S[y][x][1] = si * sg;
    }
    __syncthreads();
    // T = A1 * S
    for (int it = 0; it < 4; ++it) {
        int e = it * 256 + tid; int y = e >> 5, x = e & 31;
        float tr = 0.f, ti = 0.f;
        for (int k = 0; k < 32; ++k) {
            float ar = A1s[(y * 32 + k) * 2], ai = A1s[(y * 32 + k) * 2 + 1];
            float sr = S[k][x][0], si = S[k][x][1];
            tr += ar * sr - ai * si; ti += ar * si + ai * sr;
        }
        T[y][x][0] = tr; T[y][x][1] = ti;
    }
    __syncthreads();
    // U[:,q] = (T * B1^T) * cz(r=2)
    for (int it = 0; it < 4; ++it) {
        int e = it * 256 + tid; int y = e >> 5, x = e & 31;
        float ur = 0.f, ui = 0.f;
        for (int k = 0; k < 32; ++k) {
            float tr = T[y][k][0], ti = T[y][k][1];
            float br = B1s[(x * 32 + k) * 2], bi = B1s[(x * 32 + k) * 2 + 1];
            ur += tr * br - ti * bi; ui += tr * bi + ti * br;
        }
        float sg = czsign((y << 5) | x, 2);
        ur *= sg; ui *= sg;
        int p = (y << 5) | x;
        if (only_col0) { s0[p * 2] = ur; s0[p * 2 + 1] = ui; }
        else           { Br[q * DSTATE + p] = f2bf(ur); Bi[q * DSTATE + p] = f2bf(ui); }
    }
}

// ---------------------------------------------------------------------------
// Apply per-sample RZ diagonal, convert state to bf16 A-planes
// ---------------------------------------------------------------------------
__global__ __launch_bounds__(256) void prep_a_kernel(const float* __restrict__ ang,
                                                     const float* __restrict__ s0,
                                                     const float* __restrict__ Cre,
                                                     const float* __restrict__ Cim,
                                                     int use_s0,
                                                     unsigned short* __restrict__ Ar,
                                                     unsigned short* __restrict__ Ai) {
    __shared__ float a[WIRES];
    int b = blockIdx.x, tid = threadIdx.x;
    if (tid < WIRES) a[tid] = ang[b * WIRES + tid];
    __syncthreads();
    for (int it = 0; it < 4; ++it) {
        int q = it * 256 + tid;
        float t = 0.f;
        for (int j = 0; j < WIRES; ++j)
            t += (((q >> (9 - j)) & 1) ? 0.5f : -0.5f) * a[j];
        float sn, cs; __sincosf(t, &sn, &cs);
        float re, im;
        if (use_s0) { re = s0[q * 2]; im = s0[q * 2 + 1]; }
        else        { re = Cre[b * DSTATE + q]; im = Cim[b * DSTATE + q]; }
        float nr = re * cs - im * sn, ni = re * sn + im * cs;
        Ar[b * DSTATE + q] = f2bf(nr);
        Ai[b * DSTATE + q] = f2bf(ni);
    }
}

// ---------------------------------------------------------------------------
// Complex GEMM via WMMA bf16:  C = A (4096x1024) * B (1024x1024)
// Cr = Ar*Br + Ai*(-Bi) ; Ci = Ar*Bi + Ai*Br   (-Bi formed by register XOR)
// ---------------------------------------------------------------------------
#define GM 4096
#define GN 1024
#define GK 1024

__global__ __launch_bounds__(256) void zgemm_wmma_kernel(const unsigned short* __restrict__ Ar,
                                                         const unsigned short* __restrict__ Ai,
                                                         const unsigned short* __restrict__ Brg,
                                                         const unsigned short* __restrict__ Big,
                                                         float* __restrict__ Cr,
                                                         float* __restrict__ Ci) {
    __shared__ __attribute__((aligned(16))) unsigned short As[2][128][40];
    __shared__ __attribute__((aligned(16))) unsigned short Bs[2][64][40];
    const int tid  = threadIdx.x;
    const int lane = tid & 31;
    const int wv   = tid >> 5;
    const int wm   = wv & 3;        // M subtile (32 rows each)
    const int wn   = wv >> 2;       // N subtile (32 cols each)
    const int m0   = blockIdx.x * 128;
    const int n0   = blockIdx.y * 64;

    v8f accR[2][2], accI[2][2];
    const v8f vz = {0.f, 0.f, 0.f, 0.f, 0.f, 0.f, 0.f, 0.f};
    for (int a = 0; a < 2; ++a)
        for (int c = 0; c < 2; ++c) { accR[a][c] = vz; accI[a][c] = vz; }

    for (int k0 = 0; k0 < GK; k0 += 32) {
        // ---- stage A (128x32 of Ar and Ai) ----
        for (int p = 0; p < 2; ++p) {
            const unsigned short* src = p ? Ai : Ar;
            for (int it = 0; it < 4; ++it) {
                int g   = it * 256 + tid;
                int row = g >> 3;
                int c4  = (g & 7) * 4;
                us4 v = *(const us4*)&src[(m0 + row) * GK + k0 + c4];
                *(us4*)&As[p][row][c4] = v;
            }
        }
        // ---- stage B (32x64 of Br and Bi), transposed into LDS ----
        for (int p = 0; p < 2; ++p) {
            const unsigned short* src = p ? Big : Brg;
            for (int it = 0; it < 2; ++it) {
                int g  = it * 256 + tid;
                int kk = g >> 4;
                int n4 = (g & 15) * 4;
                us4 v = *(const us4*)&src[(k0 + kk) * GN + n0 + n4];
                for (int e = 0; e < 4; ++e)
                    Bs[p][n4 + e][kk] = v[e];
            }
        }
        __syncthreads();

        // ---- gather WMMA fragments from LDS ----
        v16bf fAr[2], fAi[2], fBr[2], fBi[2], fBni[2];
        const int arow = wm * 32 + (lane & 15);
        const int kb   = (lane < 16) ? 0 : 8;
        for (int tr = 0; tr < 2; ++tr) {
            int row = arow + tr * 16;
            v8bf lo = *(const v8bf*)&As[0][row][kb];
            v8bf hi = *(const v8bf*)&As[0][row][kb + 16];
            fAr[tr] = __builtin_shufflevector(lo, hi, 0,1,2,3,4,5,6,7,8,9,10,11,12,13,14,15);
            lo = *(const v8bf*)&As[1][row][kb];
            hi = *(const v8bf*)&As[1][row][kb + 16];
            fAi[tr] = __builtin_shufflevector(lo, hi, 0,1,2,3,4,5,6,7,8,9,10,11,12,13,14,15);
        }
        const int brow = wn * 32 + (lane & 15);
        const int klo  = (lane < 16) ? 0 : 8;
        for (int tc = 0; tc < 2; ++tc) {
            int n = brow + tc * 16;
            v8bf lo = *(const v8bf*)&Bs[0][n][klo];
            v8bf hi = *(const v8bf*)&Bs[0][n][klo + 16];
            fBr[tc] = __builtin_shufflevector(lo, hi, 0,1,2,3,4,5,6,7,8,9,10,11,12,13,14,15);
            lo = *(const v8bf*)&Bs[1][n][klo];
            hi = *(const v8bf*)&Bs[1][n][klo + 16];
            fBi[tc] = __builtin_shufflevector(lo, hi, 0,1,2,3,4,5,6,7,8,9,10,11,12,13,14,15);
            fBni[tc] = bfneg16(fBi[tc]);
        }

        // ---- 16 WMMA per wave per K-chunk ----
        for (int tr = 0; tr < 2; ++tr)
            for (int tc = 0; tc < 2; ++tc) {
                accR[tr][tc] = __builtin_amdgcn_wmma_f32_16x16x32_bf16(
                    false, fAr[tr], false, fBr[tc], (short)0, accR[tr][tc], false, false);
                accR[tr][tc] = __builtin_amdgcn_wmma_f32_16x16x32_bf16(
                    false, fAi[tr], false, fBni[tc], (short)0, accR[tr][tc], false, false);
                accI[tr][tc] = __builtin_amdgcn_wmma_f32_16x16x32_bf16(
                    false, fAr[tr], false, fBi[tc], (short)0, accI[tr][tc], false, false);
                accI[tr][tc] = __builtin_amdgcn_wmma_f32_16x16x32_bf16(
                    false, fAi[tr], false, fBr[tc], (short)0, accI[tr][tc], false, false);
            }
        __syncthreads();
    }

    // ---- epilogue: C/D layout -> global ----
    const int nlo = lane & 15;
    const int mhi = (lane < 16) ? 0 : 8;
    for (int tr = 0; tr < 2; ++tr)
        for (int tc = 0; tc < 2; ++tc) {
            int mb = m0 + wm * 32 + tr * 16 + mhi;
            int nb = n0 + wn * 32 + tc * 16 + nlo;
            for (int r = 0; r < 8; ++r) {
                Cr[(mb + r) * GN + nb] = accR[tr][tc][r];
                Ci[(mb + r) * GN + nb] = accI[tr][tc][r];
            }
        }
}

// ---------------------------------------------------------------------------
// Probability / clip kernels
// ---------------------------------------------------------------------------
__global__ __launch_bounds__(256) void probs_small_kernel(const float* __restrict__ Cre,
                                                          const float* __restrict__ Cim,
                                                          float* __restrict__ ang) {
    int idx = blockIdx.x * 256 + threadIdx.x;
    if (idx >= BATCH * WIRES) return;
    int b = idx / WIRES, j = idx - b * WIRES;
    float re = Cre[b * DSTATE + j], im = Cim[b * DSTATE + j];
    float v = (re * re + im * im) * (float)PIX;
    ang[idx] = fminf(fmaxf(v, 0.f), 1.f);
}

__global__ __launch_bounds__(256) void probs_full_kernel(const float* __restrict__ Cre,
                                                         const float* __restrict__ Cim,
                                                         float* __restrict__ out) {
    int idx = blockIdx.x * 256 + threadIdx.x;
    float re = Cre[idx], im = Cim[idx];
    float v = (re * re + im * im) * (float)PIX;
    out[idx] = fminf(fmaxf(v, 0.f), 1.f);
}

// ---------------------------------------------------------------------------
// Host launcher
// ---------------------------------------------------------------------------
extern "C" void kernel_launch(void* const* d_in, const int* in_sizes, int n_in,
                              void* d_out, int out_size, void* d_ws, size_t ws_size,
                              hipStream_t stream) {
    (void)in_sizes; (void)n_in; (void)out_size; (void)ws_size;
    const float* X = (const float*)d_in[0];          // (4096,1,32,32) f32
    const float* Wt = (const float*)d_in[1];         // (4,4,2,10,3) f32
    char* ws = (char*)d_ws;

    float* mu    = (float*)(ws + (0u));
    float* signs = (float*)(ws + (4u << 10));
    float* ang   = (float*)(ws + (8u << 10));            // 4096*10 f32
    float* V     = (float*)(ws + (256u << 10));
    float* Wbuf  = (float*)(ws + (320u << 10));
    float* s0    = (float*)(ws + (384u << 10));          // 1024 complex f32
    float* gates = (float*)(ws + (512u << 10));          // 512 KB
    float* Xc    = (float*)(ws + (1ull << 20));          // 16 MB
    float* Cov   = (float*)(ws + (17ull << 20));         // 4 MB
    unsigned short* Brm = (unsigned short*)(ws + (21ull << 20));  // 2 MB
    unsigned short* Bim = (unsigned short*)(ws + (23ull << 20));  // 2 MB
    unsigned short* Arm = (unsigned short*)(ws + (25ull << 20));  // 8 MB
    unsigned short* Aim = (unsigned short*)(ws + (33ull << 20));  // 8 MB
    float* Cre   = (float*)(ws + (41ull << 20));         // 16 MB
    float* Cim   = (float*)(ws + (57ull << 20));         // 16 MB

    // ---------------- PCA ----------------
    mean_kernel  <<<4, 256, 0, stream>>>(X, mu);
    center_kernel<<<(BATCH * PIX) / 256, 256, 0, stream>>>(X, mu, Xc);
    cov_kernel   <<<dim3(32, 32), 256, 0, stream>>>(Xc, Cov);
    initv_kernel <<<40, 256, 0, stream>>>(V);
    for (int t = 0; t < 32; ++t) {
        matvec_kernel<<<40, 256, 0, stream>>>(Cov, V, Wbuf);
        gs_kernel    <<<1, 256, 0, stream>>>(Wbuf, V);
    }
    feats_kernel    <<<160, 256, 0, stream>>>(Xc, V, ang);
    signflip_kernel <<<WIRES, 256, 0, stream>>>(ang, signs);
    applysign_kernel<<<160, 256, 0, stream>>>(ang, signs);

    // ---------------- gate matrices for all steps/layers ----------------
    build_gates_kernel<<<32, 256, 0, stream>>>(Wt, gates);

    // ---------------- circuit steps ----------------
    for (int n = 0; n < NSTEPS; ++n) {
        for (int i = 0; i < LLAYERS; ++i) {
            const float* layerg = gates + ((size_t)((n * LLAYERS + i) * 2)) * 4096;
            if (i == 0) {
                build_u_kernel<<<1, 256, 0, stream>>>(layerg, Brm, Bim, s0, 1);
            } else {
                build_u_kernel<<<DSTATE, 256, 0, stream>>>(layerg, Brm, Bim, s0, 0);
                prep_a_kernel<<<BATCH, 256, 0, stream>>>(ang, s0, Cre, Cim,
                                                         (i == 1) ? 1 : 0, Arm, Aim);
                zgemm_wmma_kernel<<<dim3(GM / 128, GN / 64), 256, 0, stream>>>(
                    Arm, Aim, Brm, Bim, Cre, Cim);
            }
        }
        if (n < NSTEPS - 1)
            probs_small_kernel<<<160, 256, 0, stream>>>(Cre, Cim, ang);
        else
            probs_full_kernel<<<(BATCH * PIX) / 256, 256, 0, stream>>>(Cre, Cim,
                                                                       (float*)d_out);
    }
}
